// TWSABlock_1005022347820
// MI455X (gfx1250) — matrix operand order repaired and checked by reading, not compiled, not run on hardware
//
#include <hip/hip_runtime.h>
#include <hip/hip_bf16.h>
#include <math.h>

// ---------------------------------------------------------------------------
// Problem constants
// ---------------------------------------------------------------------------
#define BATCH 16
#define CH    256
#define HW    4096
#define NTOK  64
#define NH    4
#define HD    64
#define NWIN  1024
#define NPIX  65536           // BATCH*HW
#define EPSF  1e-5f

typedef __attribute__((ext_vector_type(16))) _Float16 v16h;
typedef __attribute__((ext_vector_type(8)))  float    v8f;

#define DEV static __device__ __forceinline__

// ---------------------------------------------------------------------------
// CDNA5 wave32 WMMA fragment helpers (v_wmma_f32_16x16x32_f16)
//   A: 16x32 f16  lanes 0-15: M=lane,  K=0..15
//                 lanes16-31: M=lane-16,K=16..31
//   B: 32x16 f16  lanes 0-15: N=lane, K=0..15 ; lanes16-31: N=lane-16, K=16..31
//   C/D: 16x16 f32, VGPR i -> M = i + 8*hi, N = lane&15
// ---------------------------------------------------------------------------
DEV void lane_rh(int& r, int& hi) {
  int lane = threadIdx.x & 31;
  r  = lane & 15;
  hi = lane >> 4;
}

// A tile: one contiguous 16-half run per lane (K = 16*hi + j)
DEV v16h loadA_f16(const _Float16* __restrict__ src, int ld, int m0, int k0) {
  int r, hi; lane_rh(r, hi);
  const _Float16* row = src + (size_t)(m0 + r) * ld + k0 + hi * 16;
  v16h a;
#pragma unroll
  for (int j = 0; j < 16; ++j) a[j] = row[j];
  return a;
}

// B tile, K-major source: B(k,n) = src[(k0+k)*ld + n0+n]
DEV v16h loadB_f16(const _Float16* __restrict__ src, int ld, int k0, int n0) {
  int r, hi; lane_rh(r, hi);
  v16h b;
#pragma unroll
  for (int j = 0; j < 16; ++j)
    b[j] = src[(size_t)(k0 + hi * 16 + j) * ld + n0 + r];
  return b;
}

// B tile from transposed (N-major) source: one contiguous 16-half run per lane
DEV v16h loadBT_f16(const _Float16* __restrict__ src, int ld, int k0, int n0) {
  int r, hi; lane_rh(r, hi);
  const _Float16* row = src + (size_t)(n0 + r) * ld + k0 + hi * 16;
  v16h b;
#pragma unroll
  for (int j = 0; j < 16; ++j) b[j] = row[j];
  return b;
}

DEV v8f wmma(v16h a, v16h b, v8f c) {
  return __builtin_amdgcn_wmma_f32_16x16x32_f16(false, a, false, b, (short)0, c,
                                                false, false);
}

// Double-buffered K-loop: issue next fragments before consuming current ones
// so vmem/ds loads overlap wmma execution (partial s_wait instead of wait-0).
template <int K>
DEV v8f gemm_rowBT(const _Float16* __restrict__ A, int lda, int m0,
                   const _Float16* __restrict__ B, int ldb, int n0) {
  v8f acc = {};
  v16h a0 = loadA_f16(A, lda, m0, 0);
  v16h b0 = loadBT_f16(B, ldb, 0, n0);
#pragma unroll
  for (int kk = 32; kk < K; kk += 32) {
    v16h a1 = loadA_f16(A, lda, m0, kk);
    v16h b1 = loadBT_f16(B, ldb, kk, n0);
    acc = wmma(a0, b0, acc);
    a0 = a1; b0 = b1;
  }
  return wmma(a0, b0, acc);
}

template <int K>
DEV v8f gemm_rowB(const _Float16* __restrict__ A, int lda, int m0,
                  const _Float16* __restrict__ B, int ldb, int n0) {
  v8f acc = {};
  v16h a0 = loadA_f16(A, lda, m0, 0);
  v16h b0 = loadB_f16(B, ldb, 0, n0);
#pragma unroll
  for (int kk = 32; kk < K; kk += 32) {
    v16h a1 = loadA_f16(A, lda, m0, kk);
    v16h b1 = loadB_f16(B, ldb, kk, n0);
    acc = wmma(a0, b0, acc);
    a0 = a1; b0 = b1;
  }
  return wmma(a0, b0, acc);
}

// ---------------------------------------------------------------------------
// P0: convert a weight matrix f32 -> f16 (done once; L2-resident thereafter)
// ---------------------------------------------------------------------------
__global__ __launch_bounds__(256) void k_cvt(const float* __restrict__ s,
                                             _Float16* __restrict__ d, int n) {
  int i = blockIdx.x * 256 + threadIdx.x;
  if (i < n) d[i] = (_Float16)s[i];
}

// P1: transpose x NCHW f32 -> (p, C) f16 token-major
__global__ __launch_bounds__(256) void k_cvt_x(const float* __restrict__ x,
                                               _Float16* __restrict__ xt) {
  int idx = blockIdx.x * 256 + threadIdx.x;     // p*256 + c, coalesced stores
  int p = idx >> 8, c = idx & 255;
  int b = p >> 12, pp = p & 4095;
  xt[idx] = (_Float16)x[(((size_t)b * CH + c) << 12) + pp];
}

// ---------------------------------------------------------------------------
// K1: conv1 as token-major GEMM  D(p, o) = xt(p,:) @ W1^T  + fused BN1
// one 64-pixel tile per workgroup (8 waves); writes h1 NCHW f16 for dw conv
// ---------------------------------------------------------------------------
__global__ __launch_bounds__(256) void k_conv1_bn(
    const _Float16* __restrict__ xt, const _Float16* __restrict__ w1h,
    const float* __restrict__ cb,    const float* __restrict__ g,
    const float* __restrict__ bb,    const float* __restrict__ bm,
    const float* __restrict__ bv,    _Float16* __restrict__ h1) {
  int p0 = blockIdx.x * 64;
  int wv = threadIdx.x >> 5;
  int r, hi; lane_rh(r, hi);
  const _Float16* A = xt + (size_t)p0 * CH;
  __builtin_prefetch(w1h, 0, 3);
  for (int t = wv; t < 64; t += 8) {            // 4 row-tiles x 16 col-tiles
    int m0 = (t >> 4) * 16, n0 = (t & 15) * 16;
    v8f acc = gemm_rowBT<CH>(A, CH, m0, w1h, CH, n0);
#pragma unroll
    for (int i = 0; i < 8; ++i) {
      int m = m0 + hi * 8 + i, n = n0 + r;      // m: pixel, n: out channel
      int p = p0 + m;
      float inv = g[n] * rsqrtf(bv[n] + EPSF);
      float add = (cb[n] - bm[n]) * inv + bb[n];
      h1[(((size_t)(p >> 12) * CH + n) << 12) + (p & 4095)] =
          (_Float16)(acc[i] * inv + add);
    }
  }
}

// ---------------------------------------------------------------------------
// K2: depthwise 3x3 (SAME) + bias + BN2; NCHW f16 in, token-major f16 out
// ---------------------------------------------------------------------------
__global__ __launch_bounds__(256) void k_dw_bn(
    const _Float16* __restrict__ h1, const float* __restrict__ dww,
    const float* __restrict__ db,    const float* __restrict__ g,
    const float* __restrict__ bb,    const float* __restrict__ bm,
    const float* __restrict__ bv,    _Float16* __restrict__ h2t) {
  int idx = blockIdx.x * 256 + threadIdx.x;     // NCHW linear
  int w = idx & 63, h = (idx >> 6) & 63, c = (idx >> 12) & 255, b = idx >> 20;
  const _Float16* src = h1 + ((size_t)(idx >> 12) << 12);
  const float* k9 = dww + c * 9;
  float s = 0.f;
#pragma unroll
  for (int dy = -1; dy <= 1; ++dy)
#pragma unroll
    for (int dx = -1; dx <= 1; ++dx) {
      int yy = h + dy, xx = w + dx;
      if ((unsigned)yy < 64u && (unsigned)xx < 64u)
        s += (float)src[(yy << 6) + xx] * k9[(dy + 1) * 3 + (dx + 1)];
    }
  float inv = g[c] * rsqrtf(bv[c] + EPSF);
  float add = (db[c] - bm[c]) * inv + bb[c];
  int p = (b << 12) + (h << 6) + w;
  h2t[(size_t)p * CH + c] = (_Float16)(s * inv + add);
}

// ---------------------------------------------------------------------------
// K3: conv3 as token-major GEMM + bias + residual(x) -> f32 x1 in d_out (NCHW)
// ---------------------------------------------------------------------------
__global__ __launch_bounds__(256) void k_conv3_res(
    const _Float16* __restrict__ h2t, const _Float16* __restrict__ w3h,
    const float* __restrict__ cb,     const float* __restrict__ x,
    float* __restrict__ x1) {
  int p0 = blockIdx.x * 64;
  int wv = threadIdx.x >> 5;
  int r, hi; lane_rh(r, hi);
  const _Float16* A = h2t + (size_t)p0 * CH;
  __builtin_prefetch(w3h, 0, 3);
  for (int t = wv; t < 64; t += 8) {
    int m0 = (t >> 4) * 16, n0 = (t & 15) * 16;
    v8f acc = gemm_rowBT<CH>(A, CH, m0, w3h, CH, n0);
#pragma unroll
    for (int i = 0; i < 8; ++i) {
      int m = m0 + hi * 8 + i, n = n0 + r;
      int p = p0 + m;
      size_t o = (((size_t)(p >> 12) * CH + n) << 12) + (p & 4095);
      x1[o] = acc[i] + cb[n] + x[o];
    }
  }
}

// ---------------------------------------------------------------------------
// K4: windowed attention, one 8x8 window per workgroup (8 waves).
// LDS: [0,96K) raw f32 tokens then qkv f16 | [96K,128K) tok/o f16
//      [128K,160K) per-wave scores f32     | [160K,176K) per-wave probs f16
// ---------------------------------------------------------------------------
__global__ __launch_bounds__(256) void k_attn(
    const float* __restrict__ x1, const float* __restrict__ lg,
    const float* __restrict__ lb, const _Float16* __restrict__ qwh,
    const float* __restrict__ qb, const _Float16* __restrict__ pwh,
    const float* __restrict__ pb_, float* __restrict__ x2) {
  extern __shared__ char smem[];
  float*    raw  = (float*)smem;                       // 64*256*4
  _Float16* qkv  = (_Float16*)smem;                    // 64*768*2
  _Float16* tok  = (_Float16*)(smem + 98304);          // 64*256*2 (LN'd, later o)
  int tid = threadIdx.x;
  int wv  = tid >> 5;
  float*    sc = (float*)(smem + 131072) + wv * (16 * 64);
  _Float16* pr = (_Float16*)(smem + 163840) + wv * (16 * 64);
  int r, hi; lane_rh(r, hi);

  int wid = blockIdx.x;
  int b = wid >> 6, wy = (wid >> 3) & 7, wx = wid & 7;
  const float* xb = x1 + (size_t)b * CH * HW;
  __builtin_prefetch(qwh, 0, 3);

  // ---- gather window tokens (t, c) ----
  for (int i = tid; i < NTOK * CH; i += 256) {
    int t = i >> 8, c = i & 255;
    int pix = (((wy << 3) + (t >> 3)) << 6) + (wx << 3) + (t & 7);
    raw[i] = xb[(size_t)c * HW + pix];
  }
  __syncthreads();

  // ---- LayerNorm per token ----
  if (tid < NTOK) {
    const float* row = raw + tid * CH;
    float mu = 0.f;
    for (int c = 0; c < CH; ++c) mu += row[c];
    mu *= (1.f / CH);
    float var = 0.f;
    for (int c = 0; c < CH; ++c) { float d = row[c] - mu; var += d * d; }
    float rs = rsqrtf(var * (1.f / CH) + EPSF);
    for (int c = 0; c < CH; ++c)
      tok[tid * CH + c] = (_Float16)((row[c] - mu) * rs * lg[c] + lb[c]);
  }
  __syncthreads();

  // ---- QKV GEMM: 64x768 = tok(64x256) @ qw^T, +bias ----
  for (int t = wv; t < 4 * 48; t += 8) {
    int m0 = (t / 48) * 16, n0 = (t % 48) * 16;
    v8f acc = gemm_rowBT<CH>(tok, CH, m0, qwh, CH, n0);
#pragma unroll
    for (int i = 0; i < 8; ++i) {
      int m = m0 + hi * 8 + i, n = n0 + r;
      qkv[m * 768 + n] = (_Float16)(acc[i] + qb[n]);
    }
  }
  __syncthreads();

  // ---- per-head attention: 16 units (head, 16-row tile), 2 per wave ----
  for (int u = wv; u < 16; u += 8) {
    int hd = u >> 2, m0 = (u & 3) * 16;
    // scores = q @ k^T * HD^-0.5
    for (int nt = 0; nt < 4; ++nt) {
      v8f acc = gemm_rowBT<HD>(qkv + hd * 64, 768, m0,
                               qkv + 256 + hd * 64, 768, nt * 16);
#pragma unroll
      for (int i = 0; i < 8; ++i)
        sc[(hi * 8 + i) * 64 + nt * 16 + r] = acc[i] * 0.125f;
    }
    // softmax over 64 keys (one lane per row; same-wave DS ops are in order)
    if ((threadIdx.x & 31) < 16) {
      int row = threadIdx.x & 15;
      float mx = -3.0e38f;
      for (int j = 0; j < 64; ++j) mx = fmaxf(mx, sc[row * 64 + j]);
      float s = 0.f;
      for (int j = 0; j < 64; ++j) {
        float e = __expf(sc[row * 64 + j] - mx);
        s += e;
        pr[row * 64 + j] = (_Float16)e;
      }
      float is = 1.f / s;
      for (int j = 0; j < 64; ++j)
        pr[row * 64 + j] = (_Float16)((float)pr[row * 64 + j] * is);
    }
    // o = probs @ v   (into tok buffer, reused as proj input)
    for (int nt = 0; nt < 4; ++nt) {
      v8f acc = gemm_rowB<NTOK>(pr, 64, 0,
                                qkv + 512 + hd * 64, 768, nt * 16);
#pragma unroll
      for (int i = 0; i < 8; ++i)
        tok[(m0 + hi * 8 + i) * CH + hd * 64 + nt * 16 + r] = (_Float16)acc[i];
    }
  }
  __syncthreads();

  // ---- proj GEMM + bias + residual, scatter back to NCHW ----
  for (int t = wv; t < 64; t += 8) {
    int m0 = (t >> 4) * 16, n0 = (t & 15) * 16;
    v8f acc = gemm_rowBT<CH>(tok, CH, m0, pwh, CH, n0);
#pragma unroll
    for (int i = 0; i < 8; ++i) {
      int m = m0 + hi * 8 + i, n = n0 + r;
      int pix = (((wy << 3) + (m >> 3)) << 6) + (wx << 3) + (m & 7);
      size_t o = ((size_t)b * CH + n) * HW + pix;
      x2[o] = acc[i] + pb_[n] + xb[(size_t)n * HW + pix];
    }
  }
}

// ---------------------------------------------------------------------------
// K5: LayerNorm2 over channels -> f16 tokens (p, C)
// ---------------------------------------------------------------------------
__global__ __launch_bounds__(256) void k_ln2(
    const float* __restrict__ x2, const float* __restrict__ g,
    const float* __restrict__ bt, _Float16* __restrict__ tln) {
  int p = blockIdx.x * 256 + threadIdx.x;
  const float* src = x2 + ((size_t)(p >> 12) * CH * HW) + (p & 4095);
  float mu = 0.f;
  for (int c = 0; c < CH; ++c) mu += src[(size_t)c * HW];
  mu *= (1.f / CH);
  float var = 0.f;
  for (int c = 0; c < CH; ++c) { float d = src[(size_t)c * HW] - mu; var += d * d; }
  float rs = rsqrtf(var * (1.f / CH) + EPSF);
  _Float16* dst = tln + (size_t)p * CH;
  for (int c = 0; c < CH; ++c)
    dst[c] = (_Float16)((src[(size_t)c * HW] - mu) * rs * g[c] + bt[c]);
}

// ---------------------------------------------------------------------------
// K6: fused MLP: gelu(tln @ W1^T + b1) @ W2^T + b2 + residual -> d_out
// 64x1024 f16 hidden lives in 128KB LDS
// ---------------------------------------------------------------------------
__global__ __launch_bounds__(256) void k_mlp(
    const _Float16* __restrict__ tln, const _Float16* __restrict__ w1h,
    const float* __restrict__ b1,     const _Float16* __restrict__ w2h,
    const float* __restrict__ b2,     const float* __restrict__ x2,
    float* __restrict__ out) {
  extern __shared__ char smem[];
  _Float16* hid = (_Float16*)smem;            // 64 x 1024
  int p0 = blockIdx.x * 64;
  int wv = threadIdx.x >> 5;
  int r, hi; lane_rh(r, hi);
  const _Float16* A = tln + (size_t)p0 * CH;
  __builtin_prefetch(w1h, 0, 3);
  __builtin_prefetch(w2h, 0, 3);

  // GEMM1 (64x1024, K=256) + bias + exact GELU
  for (int t = wv; t < 4 * 64; t += 8) {
    int m0 = (t >> 6) * 16, n0 = (t & 63) * 16;
    v8f acc = gemm_rowBT<CH>(A, CH, m0, w1h, CH, n0);
#pragma unroll
    for (int i = 0; i < 8; ++i) {
      int m = m0 + hi * 8 + i, n = n0 + r;
      float v = acc[i] + b1[n];
      hid[m * 1024 + n] = (_Float16)(0.5f * v * (1.f + erff(v * 0.70710678f)));
    }
  }
  __syncthreads();

  // GEMM2 (64x256, K=1024) + bias + residual, scatter to NCHW
  for (int t = wv; t < 64; t += 8) {
    int m0 = (t >> 4) * 16, n0 = (t & 15) * 16;
    v8f acc = gemm_rowBT<1024>(hid, 1024, m0, w2h, 1024, n0);
#pragma unroll
    for (int i = 0; i < 8; ++i) {
      int m = m0 + hi * 8 + i, n = n0 + r;
      int p = p0 + m;
      size_t o = ((size_t)(p >> 12) * CH + n) * HW + (p & 4095);
      out[o] = acc[i] + b2[n] + x2[o];
    }
  }
}

// ---------------------------------------------------------------------------
// host launcher
// ---------------------------------------------------------------------------
extern "C" void kernel_launch(void* const* d_in, const int* in_sizes, int n_in,
                              void* d_out, int out_size, void* d_ws, size_t ws_size,
                              hipStream_t stream) {
  const float* x       = (const float*)d_in[0];
  const float* conv1_w = (const float*)d_in[1];
  const float* conv1_b = (const float*)d_in[2];
  const float* bn1_g   = (const float*)d_in[3];
  const float* bn1_b   = (const float*)d_in[4];
  const float* bn1_m   = (const float*)d_in[5];
  const float* bn1_v   = (const float*)d_in[6];
  const float* dw_w    = (const float*)d_in[7];
  const float* dw_b    = (const float*)d_in[8];
  const float* bn2_g   = (const float*)d_in[9];
  const float* bn2_b   = (const float*)d_in[10];
  const float* bn2_m   = (const float*)d_in[11];
  const float* bn2_v   = (const float*)d_in[12];
  const float* conv3_w = (const float*)d_in[13];
  const float* conv3_b = (const float*)d_in[14];
  const float* ln1_g   = (const float*)d_in[15];
  const float* ln1_b   = (const float*)d_in[16];
  const float* qkv_w   = (const float*)d_in[17];
  const float* qkv_b   = (const float*)d_in[18];
  const float* proj_w  = (const float*)d_in[19];
  const float* proj_b  = (const float*)d_in[20];
  const float* ln2_g   = (const float*)d_in[21];
  const float* ln2_b   = (const float*)d_in[22];
  const float* mlp_w1  = (const float*)d_in[23];
  const float* mlp_b1  = (const float*)d_in[24];
  const float* mlp_w2  = (const float*)d_in[25];
  const float* mlp_b2  = (const float*)d_in[26];
  float* out = (float*)d_out;

  // workspace layout (130 MB, regions reused across phases):
  //   [0,32M):    xt f16 (P1->K1)        then x2 f32 low half   (K4->K6)
  //   [32M,64M):  h1 f16 NCHW (K1->K2)   then x2 f32 high half
  //   [64M,96M):  h2t f16 (K2->K3)
  //   [96M,128M): tln f16 (K5->K6)
  //   [128M,..):  f16 weights (P0, live whole launch)
  char* ws = (char*)d_ws;
  const size_t M = 1024 * 1024;
  _Float16* xt   = (_Float16*)ws;
  _Float16* h1   = (_Float16*)(ws + 32 * M);
  float*    x2   = (float*)ws;
  _Float16* h2t  = (_Float16*)(ws + 64 * M);
  _Float16* tln  = (_Float16*)(ws + 96 * M);
  _Float16* wb   = (_Float16*)(ws + 128 * M);
  _Float16* w1h   = wb;              // 256*256
  _Float16* w3h   = wb + 65536;      // 256*256
  _Float16* qwh   = wb + 131072;     // 768*256
  _Float16* pwh   = wb + 327680;     // 256*256
  _Float16* m1h   = wb + 393216;     // 1024*256
  _Float16* m2h   = wb + 655360;     // 256*1024

  // P0: weights -> f16 (tiny, L2-resident afterwards)
  k_cvt<<<256, 256, 0, stream>>>(conv1_w, w1h, 65536);
  k_cvt<<<256, 256, 0, stream>>>(conv3_w, w3h, 65536);
  k_cvt<<<768, 256, 0, stream>>>(qkv_w,  qwh, 196608);
  k_cvt<<<256, 256, 0, stream>>>(proj_w, pwh, 65536);
  k_cvt<<<1024, 256, 0, stream>>>(mlp_w1, m1h, 262144);
  k_cvt<<<1024, 256, 0, stream>>>(mlp_w2, m2h, 262144);
  // P1: x -> token-major f16
  k_cvt_x<<<NPIX, 256, 0, stream>>>(x, xt);

  k_conv1_bn<<<NWIN, 256, 0, stream>>>(xt, w1h, conv1_b, bn1_g, bn1_b,
                                       bn1_m, bn1_v, h1);
  k_dw_bn<<<NPIX, 256, 0, stream>>>(h1, dw_w, dw_b, bn2_g, bn2_b,
                                    bn2_m, bn2_v, h2t);
  k_conv3_res<<<NWIN, 256, 0, stream>>>(h2t, w3h, conv3_b, x, out);
  k_attn<<<NWIN, 256, 180224, stream>>>(out, ln1_g, ln1_b, qwh, qkv_b,
                                        pwh, proj_b, x2);
  k_ln2<<<NPIX / 256, 256, 0, stream>>>(x2, ln2_g, ln2_b, tln);
  k_mlp<<<NWIN, 256, 131072, stream>>>(tln, m1h, mlp_b1, m2h, mlp_b2,
                                       x2, out);
}
